// FallbackSSMKernel_22368189678380
// MI455X (gfx1250) — matrix operand order repaired and checked by reading, not compiled, not run on hardware
//
#include <hip/hip_runtime.h>
#include <hip/hip_bf16.h>
#include <cstdint>
#include <cstddef>

// ---------------- problem sizes (fixed by reference) ----------------
#define Bb      2
#define Ss      4096
#define DMm     3584
#define Hh      112
#define DHh     64
#define INNERi  7168            // H*DH
#define CHUNKc  256
#define NCHUNK  16
#define Mm      (Bb*Ss)         // 8192

// ---------------- vector types ----------------
typedef __attribute__((ext_vector_type(16))) __bf16    bf16x16;
typedef __attribute__((ext_vector_type(8)))  __bf16    bf16x8;
typedef __attribute__((ext_vector_type(8)))  float     f32x8;
typedef __attribute__((ext_vector_type(4)))  uint32_t  u32x4;
typedef __attribute__((ext_vector_type(8)))  int       i32x8;
typedef __attribute__((ext_vector_type(4)))  int       i32x4;

__device__ __forceinline__ uint16_t f32_to_bf16_rne(float f) {
    uint32_t u = __float_as_uint(f);
    uint32_t r = u + 0x7FFFu + ((u >> 16) & 1u);
    return (uint16_t)(r >> 16);
}

// ---------------- f32 -> bf16 conversion ----------------
__global__ void cvt_f32_to_bf16(const float* __restrict__ in,
                                uint16_t* __restrict__ out, int n) {
    int i = blockIdx.x * blockDim.x + threadIdx.x;
    int stride = gridDim.x * blockDim.x;
    for (; i < n; i += stride) out[i] = f32_to_bf16_rne(in[i]);
}

// ---------------- TDM tile load: 128 rows x 32 bf16 cols -> LDS ----------------
// D# per CDNA5 ISA. 2D tile, data_size=2B.
// LDS padding: pad_interval=3 (16 DWORDs = one 64B row), pad_amount=3 (4 DWORDs = 16B)
// => padded LDS row stride = 80B = 40 elements (matches fragment-load layout).
#define HAVE_TDM __has_builtin(__builtin_amdgcn_tensor_load_to_lds)

__device__ __forceinline__ void tdm_load_tile_128x32(uint32_t lds_byte_off,
                                                     const uint16_t* gsrc,
                                                     uint32_t kdim_elems) {
#if HAVE_TDM
    uint64_t ga = (uint64_t)(uintptr_t)gsrc;
    u32x4 g0;
    g0[0] = 1u;                                            // count=1, user descriptor
    g0[1] = lds_byte_off;                                  // lds_addr
    g0[2] = (uint32_t)(ga & 0xFFFFFFFFu);                  // global_addr[31:0]
    g0[3] = ((uint32_t)(ga >> 32) & 0x01FFFFFFu) | (2u << 30); // [56:32] | type=2
    i32x8 g1;
    g1[0] = (int)((1u << 16) |            // data_size = 1 -> 2 bytes
                  (1u << 20) |            // pad_enable
                  (3u << 22) |            // pad_interval = 16 DWORDs (64B)
                  (3u << 25));            // pad_amount   = 4 DWORDs (16B)
    g1[1] = (int)((kdim_elems & 0xFFFFu) << 16);                 // tensor_dim0[15:0]
    g1[2] = (int)(((kdim_elems >> 16) & 0xFFFFu) | (128u << 16));// dim0 hi | tensor_dim1=128
    g1[3] = (int)(32u << 16);                                    // dim1 hi=0 | tile_dim0=32
    g1[4] = (int)(128u);                                         // tile_dim1=128, tile_dim2=0
    g1[5] = (int)kdim_elems;                                     // tensor_dim0_stride[31:0]
    g1[6] = 0;                                                   // stride hi | dim1_stride lo
    g1[7] = 0;
    i32x4 gz = {0, 0, 0, 0};
#if defined(__clang_major__) && (__clang_major__ >= 23)
    i32x8 gz8 = {0, 0, 0, 0, 0, 0, 0, 0};
    __builtin_amdgcn_tensor_load_to_lds(g0, g1, gz, gz, gz8, 0);
#else
    __builtin_amdgcn_tensor_load_to_lds(g0, g1, gz, gz, 0);
#endif
#else
    (void)lds_byte_off; (void)gsrc; (void)kdim_elems;
#endif
}

// ---------------- tiled bf16 WMMA GEMM: C[M,N] = A[M,K] * B[N,K]^T ----------------
#define BM 128
#define BN 128
#define BK 32
#define LDSS 40                        // padded LDS row stride (elements)
#define TILE_BYTES (BM * LDSS * 2)     // 10240 B per tile

template <int KDIM, int LDC>
__global__ __launch_bounds__(256)
void gemm_bf16_wmma(const uint16_t* __restrict__ A,
                    const uint16_t* __restrict__ Bw,
                    float* __restrict__ C)
{
    // [buf][A|B] double-buffered padded tiles
    __shared__ __align__(16) uint16_t sT[2][2][BM * LDSS];

    const int tid  = threadIdx.x;
    const int wave = tid >> 5;
    const int lane = tid & 31;
    const int half = lane >> 4;
    const int l16  = lane & 15;
    const int wm   = wave & 3;      // 32 rows each
    const int wn   = wave >> 2;     // 64 cols each

    const int m0 = blockIdx.y * BM;
    const int n0 = blockIdx.x * BN;

    f32x8 acc[2][4];
#pragma unroll
    for (int i = 0; i < 2; ++i)
#pragma unroll
        for (int j = 0; j < 4; ++j) {
            f32x8 z = {0.f, 0.f, 0.f, 0.f, 0.f, 0.f, 0.f, 0.f};
            acc[i][j] = z;
        }

    constexpr int NKT = KDIM / BK;

#if HAVE_TDM
    const uint32_t ldsA0 = (uint32_t)(uintptr_t)&sT[0][0][0];
    const uint32_t ldsB0 = (uint32_t)(uintptr_t)&sT[0][1][0];
    const uint32_t ldsA1 = (uint32_t)(uintptr_t)&sT[1][0][0];
    const uint32_t ldsB1 = (uint32_t)(uintptr_t)&sT[1][1][0];
    // prologue: stage 0 into buffer 0 (one wave drives the TDM)
    if (wave == 0) {
        tdm_load_tile_128x32(ldsA0, A  + (size_t)m0 * KDIM, (uint32_t)KDIM);
        tdm_load_tile_128x32(ldsB0, Bw + (size_t)n0 * KDIM, (uint32_t)KDIM);
    }
#endif

    for (int kt = 0; kt < NKT; ++kt) {
        const int buf = kt & 1;

#if HAVE_TDM
        if (wave == 0) {
            if (kt + 1 < NKT) {
                const int k1 = (kt + 1) * BK;
                const uint32_t la = (kt & 1) ? ldsA0 : ldsA1;
                const uint32_t lb = (kt & 1) ? ldsB0 : ldsB1;
                tdm_load_tile_128x32(la, A  + (size_t)m0 * KDIM + k1, (uint32_t)KDIM);
                tdm_load_tile_128x32(lb, Bw + (size_t)n0 * KDIM + k1, (uint32_t)KDIM);
                __builtin_amdgcn_s_wait_tensorcnt(2);  // stage kt complete, kt+1 in flight
            } else {
                __builtin_amdgcn_s_wait_tensorcnt(0);
            }
        }
        __syncthreads();
#else
        // fallback: cooperative copy with manual padding
        {
            const int k0 = kt * BK;
#pragma unroll
            for (int i = 0; i < 2; ++i) {
                int c    = tid * 2 + i;        // 0..511
                int row  = c >> 2;
                int col8 = c & 3;
                const uint4* sa = (const uint4*)(A  + (size_t)(m0 + row) * KDIM + k0 + col8 * 8);
                const uint4* sb = (const uint4*)(Bw + (size_t)(n0 + row) * KDIM + k0 + col8 * 8);
                *(uint4*)(&sT[buf][0][row * LDSS + col8 * 8]) = *sa;
                *(uint4*)(&sT[buf][1][row * LDSS + col8 * 8]) = *sb;
            }
            __syncthreads();
        }
#endif

        // ---- fragments ----
        // A (16x32, 16-bit): half 0 holds K {0..7,16..23}; half 1 holds {8..15,24..31}
        bf16x16 afrag[2];
#pragma unroll
        for (int ti = 0; ti < 2; ++ti) {
            int row = wm * 32 + ti * 16 + l16;
            const bf16x8* p = (const bf16x8*)(&sT[buf][0][row * LDSS]);
            bf16x8 lo = p[half];
            bf16x8 hi = p[2 + half];
            afrag[ti] = __builtin_shufflevector(lo, hi,
                0,1,2,3,4,5,6,7,8,9,10,11,12,13,14,15);
        }
        // B (32x16, 16-bit): half 0 holds K {0..7,16..23}; half 1 holds {8..15,24..31}; n = lane%16
        bf16x16 bfrag[4];
#pragma unroll
        for (int tj = 0; tj < 4; ++tj) {
            int row = wn * 64 + tj * 16 + l16;
            const bf16x8* p = (const bf16x8*)(&sT[buf][1][row * LDSS]);
            bf16x8 lo = p[half];
            bf16x8 hi = p[2 + half];
            bfrag[tj] = __builtin_shufflevector(lo, hi,
                0,1,2,3,4,5,6,7,8,9,10,11,12,13,14,15);
        }

        // ---- 8 WMMAs per wave per k-step ----
#pragma unroll
        for (int ti = 0; ti < 2; ++ti)
#pragma unroll
            for (int tj = 0; tj < 4; ++tj)
                acc[ti][tj] = __builtin_amdgcn_wmma_f32_16x16x32_bf16(
                    false, afrag[ti], false, bfrag[tj],
                    (short)0, acc[ti][tj], false, false);

        __syncthreads();
    }

    // ---- epilogue: D layout n = lane%16, m = r + 8*half ----
#pragma unroll
    for (int ti = 0; ti < 2; ++ti) {
#pragma unroll
        for (int tj = 0; tj < 4; ++tj) {
            int n  = n0 + wn * 64 + tj * 16 + l16;
            int mb = m0 + wm * 32 + ti * 16 + 8 * half;
            float* dst = C + (size_t)mb * LDC + n;
#pragma unroll
            for (int r = 0; r < 8; ++r)
                dst[r * LDC] = acc[ti][tj][r];   // compile-time offsets
        }
    }
}

// ---------------- chunked scan + sigmoid gate, emits bf16 y ----------------
__global__ void scan_gate(const float* __restrict__ xz,
                          const float* __restrict__ A_log,
                          uint16_t* __restrict__ y)
{
    int u = blockIdx.x * blockDim.x + threadIdx.x;
    const int total = Bb * CHUNKc * INNERi;
    if (u >= total) return;
    int inner = u % INNERi;
    int p     = (u / INNERi) % CHUNKc;
    int b     = u / (INNERi * CHUNKc);
    int h     = inner / DHh;

    float Ad = __expf(-fabsf(A_log[h]));
    float hs = 0.f;
#pragma unroll
    for (int c = 0; c < NCHUNK; ++c) {
        size_t row = (size_t)(b * Ss + c * CHUNKc + p);
        const float* xr = xz + row * (2 * INNERi);
        float x = xr[inner];
        float z = xr[inner + INNERi];
        hs = hs * Ad + 0.1f * x;
        float g = 1.f / (1.f + __expf(-z));
        y[row * (size_t)INNERi + inner] = f32_to_bf16_rne(hs * g);
    }
}

// ---------------- host-side orchestration ----------------
extern "C" void kernel_launch(void* const* d_in, const int* in_sizes, int n_in,
                              void* d_out, int out_size, void* d_ws, size_t ws_size,
                              hipStream_t stream)
{
    (void)in_sizes; (void)n_in; (void)out_size; (void)ws_size;

    const float* hidden = (const float*)d_in[0];   // (B,S,DM)
    const float* W_in   = (const float*)d_in[1];   // (2*INNER, DM)
    const float* W_out  = (const float*)d_in[2];   // (DM, INNER)
    const float* A_log  = (const float*)d_in[3];   // (H,)
    float*       out    = (float*)d_out;           // (B,S,DM)

    char* ws = (char*)d_ws;
    size_t off = 0;
    uint16_t* hid_bf  = (uint16_t*)(ws + off); off += (size_t)Mm * DMm * 2;
    uint16_t* win_bf  = (uint16_t*)(ws + off); off += (size_t)2 * INNERi * DMm * 2;
    uint16_t* wout_bf = (uint16_t*)(ws + off); off += (size_t)DMm * INNERi * 2;
    uint16_t* y_bf    = (uint16_t*)(ws + off); off += (size_t)Mm * INNERi * 2;
    float*    xz      = (float*)(ws + off);    off += (size_t)Mm * 2 * INNERi * 4;

    // 1) precision conversion
    cvt_f32_to_bf16<<<4096, 256, 0, stream>>>(hidden, hid_bf,  Mm * DMm);
    cvt_f32_to_bf16<<<4096, 256, 0, stream>>>(W_in,   win_bf,  2 * INNERi * DMm);
    cvt_f32_to_bf16<<<4096, 256, 0, stream>>>(W_out,  wout_bf, DMm * INNERi);

    // 2) GEMM1: xz = hidden @ W_in^T   (M=8192, N=14336, K=3584)
    dim3 g1((2 * INNERi) / BN, Mm / BM);   // (112, 64)
    gemm_bf16_wmma<DMm, 2 * INNERi><<<g1, 256, 0, stream>>>(hid_bf, win_bf, xz);

    // 3) chunked scan + sigmoid gate -> y (bf16)
    int total = Bb * CHUNKc * INNERi;
    scan_gate<<<(total + 255) / 256, 256, 0, stream>>>(xz, A_log, y_bf);

    // 4) GEMM2: out = y @ W_out^T      (M=8192, N=3584, K=7168)
    dim3 g2(DMm / BN, Mm / BM);            // (28, 64)
    gemm_bf16_wmma<INNERi, DMm><<<g2, 256, 0, stream>>>(y_bf, wout_bf, out);
}
